// MVAEdecoder_56100862820779
// MI455X (gfx1250) — matrix operand order repaired and checked by reading, not compiled, not run on hardware
//
#include <hip/hip_runtime.h>
#include <hip/hip_bf16.h>
#include <cstdint>

// ---------------------------------------------------------------------------
// Problem constants (from reference)
// ---------------------------------------------------------------------------
#define BATCH 4096
#define D3    288
#define LW    32
#define HID   512
#define OUTD  96
#define NEXP  8
#define GHID  128

// GEMM tile config
#define BM 128
#define BN 64
#define BK 32
#define SA_STRIDE 48   // bf16 elems per LDS row (96 B => all frag reads 16B aligned)
#define SB_STRIDE 48

typedef __bf16 bf16x8  __attribute__((ext_vector_type(8)));
typedef __bf16 bf16x16 __attribute__((ext_vector_type(16)));
typedef float  v8f     __attribute__((ext_vector_type(8)));

union Frag { bf16x16 v; bf16x8 h[2]; };

__device__ __forceinline__ float eluf(float v) {
    return v > 0.0f ? v : expm1f(v);
}

// ---------------------------------------------------------------------------
// CDNA5 async global->LDS copy (ASYNCcnt path), with safe fallback.
// Builtin param types (from hipcc diagnostic): pointer to GCC-style
// 4 x int vector, AS1 (printed '__device__') source / AS3 LDS destination.
// ---------------------------------------------------------------------------
typedef int v4i_gcc __attribute__((vector_size(16)));
typedef __attribute__((address_space(1))) v4i_gcc as1_v4i;
typedef __attribute__((address_space(3))) v4i_gcc as3_v4i;

#if __has_builtin(__builtin_amdgcn_global_load_async_to_lds_b128)
#define HAS_ASYNC_LDS 1
#else
#define HAS_ASYNC_LDS 0
#endif

__device__ __forceinline__ void copy16_g2lds(const __bf16* gptr, __bf16* lptr) {
#if HAS_ASYNC_LDS
    __builtin_amdgcn_global_load_async_to_lds_b128(
        (as1_v4i*)(uintptr_t)gptr,
        (as3_v4i*)(unsigned)(uintptr_t)lptr, 0, 0);
#else
    *(uint4*)lptr = *(const uint4*)gptr;
#endif
}

__device__ __forceinline__ void async_wait0() {
#if HAS_ASYNC_LDS
#if __has_builtin(__builtin_amdgcn_s_wait_asynccnt)
    __builtin_amdgcn_s_wait_asynccnt(0);
#else
    asm volatile("s_wait_asynccnt 0" ::: "memory");
#endif
#endif
}

// ---------------------------------------------------------------------------
// f32 -> bf16 weight conversion (grid-stride)
// ---------------------------------------------------------------------------
__global__ void f32_to_bf16_kernel(const float* __restrict__ in,
                                   __bf16* __restrict__ out, int n) {
    int i = blockIdx.x * blockDim.x + threadIdx.x;
    int stride = gridDim.x * blockDim.x;
    for (; i < n; i += stride) out[i] = (__bf16)in[i];
}

// ---------------------------------------------------------------------------
// Gating MLP: concat(x,z)[B,320] -> elu(128) -> elu(128) -> softmax(8)
// One block handles 4 rows; thread t computes feature t.
// ---------------------------------------------------------------------------
#define GROWS 4
__global__ __launch_bounds__(128) void gating_kernel(
    const float* __restrict__ x, const float* __restrict__ z,
    const float* __restrict__ gW1, const float* __restrict__ gb1,
    const float* __restrict__ gW2, const float* __restrict__ gb2,
    const float* __restrict__ gW3, const float* __restrict__ gb3,
    float* __restrict__ g)
{
    __shared__ float sIn[GROWS][D3 + LW];
    __shared__ float sH1[GROWS][GHID];
    __shared__ float sH2[GROWS][GHID];
    __shared__ float sLog[GROWS][NEXP];

    const int t = threadIdx.x;
    const int row0 = blockIdx.x * GROWS;

    for (int idx = t; idx < GROWS * (D3 + LW); idx += 128) {
        int r = idx / (D3 + LW);
        int c = idx - r * (D3 + LW);
        int row = row0 + r;
        sIn[r][c] = (c < D3) ? x[(size_t)row * D3 + c]
                             : z[(size_t)row * LW + (c - D3)];
    }
    __syncthreads();

    {   // g1: 320 -> 128, elu
        float acc[GROWS];
        float bb = gb1[t];
#pragma unroll
        for (int r = 0; r < GROWS; ++r) acc[r] = bb;
        const float* wrow = gW1 + (size_t)t * (D3 + LW);
        for (int i = 0; i < D3 + LW; ++i) {
            float w = wrow[i];
#pragma unroll
            for (int r = 0; r < GROWS; ++r) acc[r] = fmaf(w, sIn[r][i], acc[r]);
        }
#pragma unroll
        for (int r = 0; r < GROWS; ++r) sH1[r][t] = eluf(acc[r]);
    }
    __syncthreads();

    {   // g2: 128 -> 128, elu
        float acc[GROWS];
        float bb = gb2[t];
#pragma unroll
        for (int r = 0; r < GROWS; ++r) acc[r] = bb;
        const float* wrow = gW2 + (size_t)t * GHID;
        for (int i = 0; i < GHID; ++i) {
            float w = wrow[i];
#pragma unroll
            for (int r = 0; r < GROWS; ++r) acc[r] = fmaf(w, sH1[r][i], acc[r]);
        }
#pragma unroll
        for (int r = 0; r < GROWS; ++r) sH2[r][t] = eluf(acc[r]);
    }
    __syncthreads();

    if (t < GROWS * NEXP) {  // g3 logits: 128 -> 8
        int r = t >> 3, e = t & 7;
        float acc = gb3[e];
        const float* wrow = gW3 + (size_t)e * GHID;
        for (int i = 0; i < GHID; ++i) acc = fmaf(wrow[i], sH2[r][i], acc);
        sLog[r][e] = acc;
    }
    __syncthreads();

    if (t < GROWS * NEXP) {  // softmax over 8 experts
        int r = t >> 3, e = t & 7;
        float mx = sLog[r][0];
#pragma unroll
        for (int j = 1; j < NEXP; ++j) mx = fmaxf(mx, sLog[r][j]);
        float sum = 0.0f;
#pragma unroll
        for (int j = 0; j < NEXP; ++j) sum += __expf(sLog[r][j] - mx);
        g[(size_t)(row0 + r) * NEXP + e] = __expf(sLog[r][e] - mx) / sum;
    }
}

// ---------------------------------------------------------------------------
// Blended-expert GEMM, expanded-K WMMA, double-buffered LDS, async B copies.
//   out[b,o] = act( sum_e sum_i W[e,o,i]*(g[b,e]*src[b,i]) + sum_e g[b,e]*bias[e,o] )
// Tile: BM=128 x BN=64 x BK=32, 256 threads = 8 waves (4x2), wave tile 32x32.
// All segment widths are multiples of 32 => each K-block has uniform (e, segment).
// ---------------------------------------------------------------------------
template <int IN, int OUT, int LD1, int LD2, bool ELU>
__global__ __launch_bounds__(256) void moe_gemm_kernel(
    const float* __restrict__ src1,               // first concat segment [B,LD1]
    const float* __restrict__ src2,               // second segment (z) or null
    const float* __restrict__ g,                  // [B, 8]
    const __bf16* __restrict__ Wb,                // [E][OUT][IN] bf16
    const float* __restrict__ bias,               // [E][OUT]
    float* __restrict__ out)                      // [B][OUT]
{
    __shared__ __bf16 sA[2][BM * SA_STRIDE];
    __shared__ __bf16 sB[2][BN * SB_STRIDE];
    __shared__ float  sG[BM][NEXP];
    __shared__ float  sBias[NEXP][BN];

    const int tid = threadIdx.x;
    const int mb0 = blockIdx.x * BM;
    const int nb0 = blockIdx.y * BN;

    // cooperative load assignments
    const int am  = tid >> 1;           // A: row 0..127
    const int akk = (tid & 1) * 16;     // A: 16 k-elems each
    const int bn  = tid & 63;           // B: col 0..63
    const int bkk = (tid >> 6) * 8;     // B: 8 k-elems each
    const int nglob = nb0 + bn;

    // stage gate rows for this M-tile
    for (int idx = tid; idx < BM * NEXP; idx += 256)
        sG[idx >> 3][idx & 7] = g[(size_t)(mb0 + (idx >> 3)) * NEXP + (idx & 7)];
    // stage bias tile (zero-pad cols beyond OUT)
    for (int idx = tid; idx < NEXP * BN; idx += 256) {
        int e = idx / BN, n = idx - e * BN, col = nb0 + n;
        sBias[e][n] = (col < OUT) ? bias[(size_t)e * OUT + col] : 0.0f;
    }
    // pre-zero ragged-N B rows once (both buffers); never written again
    if (OUT % BN != 0 && nglob >= OUT) {
        uint4 zz = make_uint4(0u, 0u, 0u, 0u);
        *(uint4*)&sB[0][bn * SB_STRIDE + bkk] = zz;
        *(uint4*)&sB[1][bn * SB_STRIDE + bkk] = zz;
    }
    __syncthreads();   // sG/sBias/zeros visible before first tile load

    const int wid  = tid >> 5, lane = tid & 31;
    const int wm   = wid >> 1, wn   = wid & 1;       // 4x2 wave grid
    const int lm   = lane & 15;
    const int klo  = (lane < 16) ? 0 : 8;            // A-frag K offset
    const int kbas = (lane < 16) ? 0 : 8;            // B-frag K offset

    v8f acc[2][2] = {};

    // incrementally maintained per-step state
    const float* rowA1 = src1 + (size_t)(mb0 + am) * LD1 + akk;
    const float* rowA2 = (LD2 > 0) ? (src2 + (size_t)(mb0 + am) * LD2 + akk)
                                   : rowA1;
    const __bf16* wbase = Wb + (size_t)nglob * IN + bkk;   // used only if nglob<OUT

    const float* aptr = rowA1;     // A source for the tile about to be loaded
    size_t woff = 0;               // e*OUT*IN + i0 for that tile
    int e_cur = 0, i0_cur = 0;

    auto load_tiles = [&](int buf) {
        // ---- A tile: bf16( g[row,e] * src[row, i] ) ----
        const float gs = sG[am][e_cur];
        const float4* p4 = (const float4*)aptr;
        __bf16* dst = &sA[buf][am * SA_STRIDE + akk];
#pragma unroll
        for (int j = 0; j < 4; ++j) {
            float4 f = p4[j];
            dst[4 * j + 0] = (__bf16)(gs * f.x);
            dst[4 * j + 1] = (__bf16)(gs * f.y);
            dst[4 * j + 2] = (__bf16)(gs * f.z);
            dst[4 * j + 3] = (__bf16)(gs * f.w);
        }
        // ---- B tile: W[e, n, i0 + kk] (contiguous in k) via async copy ----
        if (OUT % BN == 0 || nglob < OUT)
            copy16_g2lds(wbase + woff, &sB[buf][bn * SB_STRIDE + bkk]);
    };
    auto advance = [&]() {
        i0_cur += BK;
        aptr   += BK;
        woff   += BK;
        if (LD2 > 0 && i0_cur == LD1) aptr = rowA2;           // hop to z segment
        if (i0_cur == IN) {                                    // next expert
            i0_cur = 0;
            ++e_cur;
            aptr = rowA1;
            woff += (size_t)OUT * IN - IN;
        }
    };
    auto compute = [&](int buf) {
        const __bf16* A = sA[buf];
        const __bf16* B = sB[buf];
        Frag a[2], b[2];
#pragma unroll
        for (int mt = 0; mt < 2; ++mt) {
            const int row = wm * 32 + mt * 16 + lm;
            a[mt].h[0] = *(const bf16x8*)&A[row * SA_STRIDE + klo];
            a[mt].h[1] = *(const bf16x8*)&A[row * SA_STRIDE + 16 + klo];
        }
#pragma unroll
        for (int nt = 0; nt < 2; ++nt) {
            const int col = wn * 32 + nt * 16 + lm;
            b[nt].h[0] = *(const bf16x8*)&B[col * SB_STRIDE + kbas];
            b[nt].h[1] = *(const bf16x8*)&B[col * SB_STRIDE + kbas + 16];
        }
#pragma unroll
        for (int mt = 0; mt < 2; ++mt)
#pragma unroll
            for (int nt = 0; nt < 2; ++nt)
                acc[mt][nt] = __builtin_amdgcn_wmma_f32_16x16x32_bf16(
                    false, a[mt].v, false, b[nt].v,
                    (short)0, acc[mt][nt], false, false);
    };

    constexpr int STEPS = NEXP * (IN / BK);

    // prologue: fill buffer 0
    load_tiles(0);
    advance();

    int buf = 0;
    for (int s = 0; s < STEPS; ++s) {
        async_wait0();       // this wave's async B copy for `buf` has landed
        __syncthreads();     // all waves' A stores + B copies for `buf` visible
        if (s + 1 < STEPS) { // overlap next tile's loads with this tile's WMMAs
            load_tiles(buf ^ 1);
            advance();
        }
        compute(buf);
        buf ^= 1;
    }

    // ---- epilogue: blended bias + activation + store ----
    const int moff = (lane < 16) ? 0 : 8;
#pragma unroll
    for (int mt = 0; mt < 2; ++mt) {
#pragma unroll
        for (int nt = 0; nt < 2; ++nt) {
            const int nl  = wn * 32 + nt * 16 + lm;
            const int col = nb0 + nl;
            if (OUT % BN == 0 || col < OUT) {
#pragma unroll
                for (int v = 0; v < 8; ++v) {
                    const int ml = wm * 32 + mt * 16 + v + moff;
                    float bb = 0.0f;
#pragma unroll
                    for (int e = 0; e < NEXP; ++e)
                        bb = fmaf(sG[ml][e], sBias[e][nl], bb);
                    float val = acc[mt][nt][v] + bb;
                    if (ELU) val = eluf(val);
                    out[(size_t)(mb0 + ml) * OUT + col] = val;
                }
            }
        }
    }
}

// ---------------------------------------------------------------------------
// Launch
// ---------------------------------------------------------------------------
extern "C" void kernel_launch(void* const* d_in, const int* in_sizes, int n_in,
                              void* d_out, int out_size, void* d_ws, size_t ws_size,
                              hipStream_t stream) {
    (void)in_sizes; (void)n_in; (void)out_size; (void)ws_size;

    const float* x   = (const float*)d_in[0];
    const float* z   = (const float*)d_in[1];
    const float* W0  = (const float*)d_in[2];
    const float* b0  = (const float*)d_in[3];
    const float* W1  = (const float*)d_in[4];
    const float* b1  = (const float*)d_in[5];
    const float* W2  = (const float*)d_in[6];
    const float* b2  = (const float*)d_in[7];
    const float* W3  = (const float*)d_in[8];
    const float* b3  = (const float*)d_in[9];
    const float* gW1 = (const float*)d_in[10];
    const float* gb1 = (const float*)d_in[11];
    const float* gW2 = (const float*)d_in[12];
    const float* gb2 = (const float*)d_in[13];
    const float* gW3 = (const float*)d_in[14];
    const float* gb3 = (const float*)d_in[15];
    float* out = (float*)d_out;

    // workspace carve-up (256B aligned)
    char* ws = (char*)d_ws;
    size_t off = 0;
    auto carve = [&](size_t bytes) {
        size_t r = off;
        off = (off + bytes + 255) & ~(size_t)255;
        return r;
    };
    const int nW0 = NEXP * HID * (D3 + LW);      // 8*512*320
    const int nW1 = NEXP * HID * (HID + LW);     // 8*512*544
    const int nW2 = nW1;
    const int nW3 = NEXP * OUTD * HID;           // 8*96*512

    float*  g_buf = (float*)(ws + carve((size_t)BATCH * NEXP * 4));
    float*  hA    = (float*)(ws + carve((size_t)BATCH * HID * 4));
    float*  hB    = (float*)(ws + carve((size_t)BATCH * HID * 4));
    __bf16* W0b   = (__bf16*)(ws + carve((size_t)nW0 * 2));
    __bf16* W1b   = (__bf16*)(ws + carve((size_t)nW1 * 2));
    __bf16* W2b   = (__bf16*)(ws + carve((size_t)nW2 * 2));
    __bf16* W3b   = (__bf16*)(ws + carve((size_t)nW3 * 2));

    // 1) weight conversion f32 -> bf16
    f32_to_bf16_kernel<<<2048, 256, 0, stream>>>(W0, W0b, nW0);
    f32_to_bf16_kernel<<<2048, 256, 0, stream>>>(W1, W1b, nW1);
    f32_to_bf16_kernel<<<2048, 256, 0, stream>>>(W2, W2b, nW2);
    f32_to_bf16_kernel<<<2048, 256, 0, stream>>>(W3, W3b, nW3);

    // 2) gating -> g[B,8]
    gating_kernel<<<BATCH / GROWS, 128, 0, stream>>>(
        x, z, gW1, gb1, gW2, gb2, gW3, gb3, g_buf);

    // 3) blended layers (expanded-K WMMA GEMMs)
    moe_gemm_kernel<D3 + LW, HID, D3, LW, true>
        <<<dim3(BATCH / BM, HID / BN), 256, 0, stream>>>(
            x, z, g_buf, W0b, b0, hA);
    moe_gemm_kernel<HID + LW, HID, HID, LW, true>
        <<<dim3(BATCH / BM, HID / BN), 256, 0, stream>>>(
            hA, z, g_buf, W1b, b1, hB);
    moe_gemm_kernel<HID + LW, HID, HID, LW, true>
        <<<dim3(BATCH / BM, HID / BN), 256, 0, stream>>>(
            hB, z, g_buf, W2b, b2, hA);
    moe_gemm_kernel<HID, OUTD, HID, 0, false>
        <<<dim3(BATCH / BM, (OUTD + BN - 1) / BN), 256, 0, stream>>>(
            hA, nullptr, g_buf, W3b, b3, out);
}
